// MultiQueryAttention_78365973283071
// MI455X (gfx1250) — matrix operand order repaired and checked by reading, not compiled, not run on hardware
//
#include <hip/hip_runtime.h>
#include <hip/hip_bf16.h>
#include <math.h>

// ---------------------------------------------------------------------------
// MQA forward for gfx1250 (MI455X): B=2, T=2048, DIM=2048, H=16, D=128 (MQA).
// BF16 WMMA (f32 accum) for all GEMMs + flash attention.
// TDM (tensor_load_to_lds) stages the K tile; s_wait_tensorcnt/dscnt sync.
// ---------------------------------------------------------------------------

constexpr int kB   = 2;
constexpr int kT   = 2048;
constexpr int kDim = 2048;
constexpr int kH   = 16;
constexpr int kD   = 128;         // head dim
constexpr int kM   = kB * kT;     // 4096 tokens

typedef __attribute__((ext_vector_type(16))) __bf16 v16bf;
typedef __attribute__((ext_vector_type(8)))  __bf16 v8bf;
typedef __attribute__((ext_vector_type(8)))  float  v8f;
typedef __attribute__((ext_vector_type(4)))  unsigned int v4u;
typedef __attribute__((ext_vector_type(8)))  int v8i;
typedef __attribute__((ext_vector_type(4)))  int v4i;

#if defined(__HIP_DEVICE_COMPILE__) && \
    __has_builtin(__builtin_amdgcn_tensor_load_to_lds) && \
    __has_builtin(__builtin_amdgcn_s_wait_tensorcnt)
#define MQA_USE_TDM 1
#else
#define MQA_USE_TDM 0
#endif

__device__ __forceinline__ __bf16 f2bf(float f) {
  union { float f; unsigned u; } v; v.f = f;
  unsigned r = v.u + 0x7FFFu + ((v.u >> 16) & 1u);   // round-to-nearest-even
  union { unsigned short us; __bf16 b; } c; c.us = (unsigned short)(r >> 16);
  return c.b;
}

// ---------------------------------------------------------------------------
// Generic tiled GEMM:  C[M,N] = A[M,K] * B[K,N] + bias[N]
// Block tile 128x128, K-step 32. 256 threads = 8 waves, each wave owns a
// 32x64 sub-tile = 2x4 grid of 16x16 WMMA tiles. A and B are converted to
// bf16 while staging into LDS; B is stored transposed for contiguous
// B-fragment reads.
// ---------------------------------------------------------------------------
template <bool A_IS_BF16>
__global__ __launch_bounds__(256)
void gemm_bias_kernel(const void* __restrict__ Aptr,
                      const float* __restrict__ Bmat,
                      const float* __restrict__ bias,
                      float* __restrict__ C,
                      int M, int N, int K) {
  __shared__ alignas(16) __bf16 As[128 * 40];   // row m, padded stride 40
  __shared__ alignas(16) __bf16 Bt[128 * 40];   // row n (transposed), stride 40

  const int tid  = threadIdx.x;
  const int lane = tid & 31;
  const int w    = tid >> 5;          // wave 0..7
  const int wm   = (w & 3) * 32;      // wave M offset inside 128 tile
  const int wn   = (w >> 2) * 64;     // wave N offset inside 128 tile
  const int half = lane >> 4;         // 0 / 1 (lane half)
  const int l16  = lane & 15;

  const int m0 = blockIdx.y * 128;
  const int n0 = blockIdx.x * 128;

  v8f acc[2][4];
#pragma unroll
  for (int i = 0; i < 2; ++i)
#pragma unroll
    for (int j = 0; j < 4; ++j)
#pragma unroll
      for (int r = 0; r < 8; ++r) acc[i][j][r] = 0.0f;

  for (int k0 = 0; k0 < K; k0 += 32) {
    // ---- stage A tile: 128 rows x 32 k-cols (convert f32 -> bf16) ----
    {
      const int row = tid & 127;
      const int cb  = (tid >> 7) * 16;          // 0 or 16
      if constexpr (A_IS_BF16) {
        const __bf16* A  = (const __bf16*)Aptr;
        const __bf16* src = A + (size_t)(m0 + row) * K + (k0 + cb);
        v8bf x0 = *(const v8bf*)(src);
        v8bf x1 = *(const v8bf*)(src + 8);
        *(v8bf*)&As[row * 40 + cb]     = x0;
        *(v8bf*)&As[row * 40 + cb + 8] = x1;
        if (k0 + 32 < K) __builtin_prefetch(src + 32, 0, 1);   // global_prefetch_b8
      } else {
        const float* A   = (const float*)Aptr;
        const float* src = A + (size_t)(m0 + row) * K + (k0 + cb);
        v8bf o0, o1;
#pragma unroll
        for (int i = 0; i < 8; ++i) o0[i] = f2bf(src[i]);
#pragma unroll
        for (int i = 0; i < 8; ++i) o1[i] = f2bf(src[8 + i]);
        *(v8bf*)&As[row * 40 + cb]     = o0;
        *(v8bf*)&As[row * 40 + cb + 8] = o1;
        if (k0 + 32 < K) __builtin_prefetch(src + 32, 0, 1);   // global_prefetch_b8
      }
    }
    // ---- stage B tile transposed: Bt[n][k], 32 k-rows x 128 n-cols ----
    {
      const int kk = tid & 31;
      const int cb = (tid >> 5) * 16;           // 0..112
      const float* src = Bmat + (size_t)(k0 + kk) * N + (n0 + cb);
#pragma unroll
      for (int i = 0; i < 16; ++i) Bt[(cb + i) * 40 + kk] = f2bf(src[i]);
      if (k0 + 32 < K) __builtin_prefetch(src + (size_t)32 * N, 0, 1);
    }
    __syncthreads();

    // ---- A fragments (16x32, ISA layout) ----
    v16bf afrag[2];
#pragma unroll
    for (int mt = 0; mt < 2; ++mt) {
      const __bf16* p = &As[(wm + mt * 16 + l16) * 40 + half * 8];
      v8bf lo = *(const v8bf*)p;
      v8bf hi = *(const v8bf*)(p + 16);
#pragma unroll
      for (int i = 0; i < 8; ++i) { afrag[mt][i] = lo[i]; afrag[mt][8 + i] = hi[i]; }
    }
    // ---- B fragments (32x16) + WMMAs ----
#pragma unroll
    for (int nt = 0; nt < 4; ++nt) {
      const __bf16* p = &Bt[(wn + nt * 16 + l16) * 40 + half * 8];
      v8bf lo = *(const v8bf*)p;
      v8bf hi = *(const v8bf*)(p + 16);
      v16bf bfrag;
#pragma unroll
      for (int i = 0; i < 8; ++i) { bfrag[i] = lo[i]; bfrag[8 + i] = hi[i]; }
#pragma unroll
      for (int mt = 0; mt < 2; ++mt) {
        acc[mt][nt] = __builtin_amdgcn_wmma_f32_16x16x32_bf16(
            false, afrag[mt], false, bfrag, (short)0, acc[mt][nt], false, false);
      }
    }
    __syncthreads();
  }

  // ---- epilogue: C/D layout -> global, add bias ----
#pragma unroll
  for (int mt = 0; mt < 2; ++mt) {
#pragma unroll
    for (int nt = 0; nt < 4; ++nt) {
      const int col = n0 + wn + nt * 16 + l16;
      const float bv = bias[col];
#pragma unroll
      for (int r = 0; r < 8; ++r) {
        const int row = m0 + wm + mt * 16 + r + 8 * half;
        C[(size_t)row * N + col] = acc[mt][nt][r] + bv;
      }
    }
  }
}

// ---------------------------------------------------------------------------
// RoPE for Q: q_f32 [B*T, H*D] -> qb bf16 [B,H,T,D] (rotate-half pairing d, d+64)
// ---------------------------------------------------------------------------
__global__ __launch_bounds__(256)
void rope_q_kernel(const float* __restrict__ qf, __bf16* __restrict__ qb) {
  const int idx = blockIdx.x * 256 + threadIdx.x;   // kB*kT*kH*64 = 2^22
  const int jj = idx & 63;
  const int h  = (idx >> 6) & 15;
  const int t  = (idx >> 10) & (kT - 1);
  const int b  = idx >> 21;
  const float invf  = __expf(-(float)jj * (9.210340371976184f / 64.0f)); // 10000^(-j/64)
  const float ang = (float)t * invf;
  const float c = __cosf(ang), s = __sinf(ang);
  const float* src = qf + ((size_t)(b * kT + t)) * kDim + h * kD;
  __bf16* dst = qb + ((size_t)(b * kH + h) * kT + t) * kD;
  const float x0 = src[jj], x1 = src[jj + 64];
  dst[jj]      = f2bf(x0 * c - x1 * s);
  dst[jj + 64] = f2bf(x1 * c + x0 * s);
}

// RoPE for K: k_f32 [B*T, D] -> kb bf16 same layout
__global__ __launch_bounds__(256)
void rope_k_kernel(const float* __restrict__ kf, __bf16* __restrict__ kb) {
  const int idx = blockIdx.x * 256 + threadIdx.x;   // kB*kT*64
  const int jj = idx & 63;
  const int t  = (idx >> 6) & (kT - 1);
  const int b  = idx >> 17;
  const float invf = __expf(-(float)jj * (9.210340371976184f / 64.0f));
  const float ang = (float)t * invf;
  const float c = __cosf(ang), s = __sinf(ang);
  const float* src = kf + ((size_t)(b * kT + t)) * kD;
  __bf16* dst = kb + ((size_t)(b * kT + t)) * kD;
  const float x0 = src[jj], x1 = src[jj + 64];
  dst[jj]      = f2bf(x0 * c - x1 * s);
  dst[jj + 64] = f2bf(x1 * c + x0 * s);
}

__global__ __launch_bounds__(256)
void cast_v_kernel(const float* __restrict__ vf, __bf16* __restrict__ vb) {
  const int idx = blockIdx.x * 256 + threadIdx.x;
  vb[idx] = f2bf(vf[idx]);
}

// ---------------------------------------------------------------------------
// Flash attention (causal, MQA). Block = (q-tile of 128 rows, head h, batch b).
// 8 waves; each wave owns 16 q rows. K tile (64x128) staged by the Tensor
// Data Mover (wave 0 issues one TENSOR_LOAD_TO_LDS per kv tile, with LDS row
// padding 256B->272B via pad_interval/pad_amount); transposed V tile staged
// cooperatively; P re-shaped via per-wave LDS scratch.
// ---------------------------------------------------------------------------
__global__ __launch_bounds__(256)
void flash_mqa_kernel(const __bf16* __restrict__ Qb,   // [B,H,T,D]
                      const __bf16* __restrict__ Kb,   // [B,T,D]
                      const __bf16* __restrict__ Vb,   // [B,T,D]
                      __bf16* __restrict__ Ob) {       // [B,T,H*D]
  __shared__ alignas(16) __bf16 Ks[64 * 136];        // K rows, stride 136 (272B)
  __shared__ alignas(16) __bf16 Vt[128 * 72];        // V transposed [d][kv], stride 72
  __shared__ alignas(16) __bf16 Ps[8 * 16 * 72];     // per-wave 16x64 P tile

  const int tid  = threadIdx.x;
  const int lane = tid & 31;
  const int w    = tid >> 5;
  const int half = lane >> 4;
  const int l16  = lane & 15;

  const int qt = blockIdx.x;   // q tile (0..15)
  const int h  = blockIdx.y;   // head
  const int b  = blockIdx.z;   // batch

  const int q0 = qt * 128 + w * 16;   // wave's first q row

  // Q fragments: 4 K-chunks of 32 over D=128
  v16bf qfrag[4];
  {
    const __bf16* qsrc = Qb + ((size_t)(b * kH + h) * kT + q0 + l16) * kD;
#pragma unroll
    for (int c = 0; c < 4; ++c) {
      v8bf lo = *(const v8bf*)(qsrc + c * 32 + half * 8);
      v8bf hi = *(const v8bf*)(qsrc + c * 32 + 16 + half * 8);
#pragma unroll
      for (int i = 0; i < 8; ++i) { qfrag[c][i] = lo[i]; qfrag[c][8 + i] = hi[i]; }
    }
  }

  v8f out[8];
#pragma unroll
  for (int d = 0; d < 8; ++d)
#pragma unroll
    for (int r = 0; r < 8; ++r) out[d][r] = 0.0f;

  float mrow[8], lrow[8];
#pragma unroll
  for (int r = 0; r < 8; ++r) { mrow[r] = -3.0e38f; lrow[r] = 0.0f; }

  const float scale = 0.08838834764831845f;   // 1/sqrt(128)
  const int jend = qt * 128 + 128;

  for (int j = 0; j < jend; j += 64) {
#if MQA_USE_TDM
    // ---- wave 0 issues TDM DMA: K tile (64 rows x 128 bf16) -> Ks ----
    if (w == 0) {
      const unsigned long long ga =
          (unsigned long long)(const void*)(Kb + ((size_t)b * kT + j) * kD);
      const unsigned ldsa = (unsigned)(unsigned long long)(void*)&Ks[0];
      // D# group 0: count=1 | lds_addr | global_addr[56:0] | type=2
      v4u g0 = { 1u, ldsa,
                 (unsigned)(ga & 0xffffffffu),
                 (unsigned)((ga >> 32) & 0x01ffffffu) | (2u << 30) };
      // D# group 1: data_size=2B, pad_enable, pad_interval=64DW, pad_amount=4DW,
      //             tensor 128x64, tile 128x64, dim0 stride 128
      v8i g1;
      g1[0] = (int)((1u << 16) | (1u << 20) | (5u << 22) | (3u << 25));
      g1[1] = (int)(128u << 16);           // tensor_dim0 = 128 (lo16)
      g1[2] = (int)(64u << 16);            // tensor_dim0 hi=0 | tensor_dim1 = 64
      g1[3] = (int)(128u << 16);           // tensor_dim1 hi=0 | tile_dim0 = 128
      g1[4] = (int)(64u);                  // tile_dim1 = 64 | tile_dim2 = 0
      g1[5] = (int)(128u);                 // tensor_dim0_stride = 128
      g1[6] = 0;
      g1[7] = 0;
      v4i gz = { 0, 0, 0, 0 };
#if defined(__clang_major__) && (__clang_major__ >= 23)
      v8i gz8 = { 0, 0, 0, 0, 0, 0, 0, 0 };
      __builtin_amdgcn_tensor_load_to_lds(g0, g1, gz, gz, gz8, 0);
#else
      __builtin_amdgcn_tensor_load_to_lds(g0, g1, gz, gz, 0);
#endif
    }
#endif
    // ---- cooperative staging of transposed V (and K on non-TDM path) ----
#pragma unroll
    for (int it = 0; it < 4; ++it) {
      const int idx = tid + it * 256;          // 0..1023
      const int row = idx >> 4;                // kv row 0..63
      const int c8  = (idx & 15) * 8;          // d col
      v8bf vv = *(const v8bf*)(Vb + ((size_t)b * kT + j + row) * kD + c8);
#pragma unroll
      for (int i = 0; i < 8; ++i) Vt[(c8 + i) * 72 + row] = vv[i];
#if !MQA_USE_TDM
      v8bf kv = *(const v8bf*)(Kb + ((size_t)b * kT + j + row) * kD + c8);
      *(v8bf*)&Ks[row * 136 + c8] = kv;
#endif
    }
#if MQA_USE_TDM
    if (w == 0) __builtin_amdgcn_s_wait_tensorcnt(0);
#endif
    __syncthreads();

    // ---- S = Q @ K^T, 4 n-subtiles of 16 kv cols ----
    v8f s[4];
#pragma unroll
    for (int nt = 0; nt < 4; ++nt) {
      v8f accS;
#pragma unroll
      for (int r = 0; r < 8; ++r) accS[r] = 0.0f;
#pragma unroll
      for (int c = 0; c < 4; ++c) {
        const __bf16* p = &Ks[(nt * 16 + l16) * 136 + c * 32 + half * 8];
        v8bf lo = *(const v8bf*)p;
        v8bf hi = *(const v8bf*)(p + 16);
        v16bf kf;
#pragma unroll
        for (int i = 0; i < 8; ++i) { kf[i] = lo[i]; kf[8 + i] = hi[i]; }
        accS = __builtin_amdgcn_wmma_f32_16x16x32_bf16(
            false, qfrag[c], false, kf, (short)0, accS, false, false);
      }
      s[nt] = accS;
    }

    // ---- scale, causal mask, online softmax row stats ----
    float mnew[8];
#pragma unroll
    for (int r = 0; r < 8; ++r) {
      const int qg = q0 + r + 8 * half;
      float mx = -3.0e38f;
#pragma unroll
      for (int nt = 0; nt < 4; ++nt) {
        const int kvc = j + nt * 16 + l16;
        float v = s[nt][r] * scale;
        v = (kvc <= qg) ? v : -3.0e38f;
        s[nt][r] = v;
        mx = fmaxf(mx, v);
      }
      for (int msk = 1; msk < 16; msk <<= 1)
        mx = fmaxf(mx, __shfl_xor(mx, msk, 32));
      mnew[r] = fmaxf(mrow[r], mx);
    }

    float alpha[8];
#pragma unroll
    for (int r = 0; r < 8; ++r) {
      const float a = __expf(mrow[r] - mnew[r]);
      alpha[r] = a;
      float sum = 0.0f;
#pragma unroll
      for (int nt = 0; nt < 4; ++nt) {
        const float p = __expf(s[nt][r] - mnew[r]);
        s[nt][r] = p;
        sum += p;
      }
      for (int msk = 1; msk < 16; msk <<= 1)
        sum += __shfl_xor(sum, msk, 32);
      lrow[r] = lrow[r] * a + sum;
      mrow[r] = mnew[r];
    }
#pragma unroll
    for (int d = 0; d < 8; ++d)
#pragma unroll
      for (int r = 0; r < 8; ++r) out[d][r] *= alpha[r];

    // ---- P: C/D layout -> per-wave LDS -> A-fragment layout ----
    {
      __bf16* pw = &Ps[w * 16 * 72];
#pragma unroll
      for (int nt = 0; nt < 4; ++nt)
#pragma unroll
        for (int r = 0; r < 8; ++r)
          pw[(r + 8 * half) * 72 + nt * 16 + l16] = f2bf(s[nt][r]);
    }
    // wave-private LDS region: only need our own DS ops complete (CDNA5 split counter)
    asm volatile("s_wait_dscnt 0" ::: "memory");

    // ---- out += P @ V ----
#pragma unroll
    for (int cc = 0; cc < 2; ++cc) {
      const __bf16* pr = &Ps[w * 16 * 72] + l16 * 72 + cc * 32 + half * 8;
      v8bf lo = *(const v8bf*)pr;
      v8bf hi = *(const v8bf*)(pr + 16);
      v16bf pf;
#pragma unroll
      for (int i = 0; i < 8; ++i) { pf[i] = lo[i]; pf[8 + i] = hi[i]; }
#pragma unroll
      for (int d = 0; d < 8; ++d) {
        const __bf16* vp = &Vt[(d * 16 + l16) * 72 + cc * 32 + half * 8];
        v8bf vlo = *(const v8bf*)vp;
        v8bf vhi = *(const v8bf*)(vp + 16);
        v16bf vf;
#pragma unroll
        for (int i = 0; i < 8; ++i) { vf[i] = vlo[i]; vf[8 + i] = vhi[i]; }
        out[d] = __builtin_amdgcn_wmma_f32_16x16x32_bf16(
            false, pf, false, vf, (short)0, out[d], false, false);
      }
    }
    __syncthreads();
  }

  // ---- normalize + store to [B,T,H*D] (bf16 for the O projection) ----
#pragma unroll
  for (int d = 0; d < 8; ++d) {
#pragma unroll
    for (int r = 0; r < 8; ++r) {
      const int qg = q0 + r + 8 * half;
      const float val = out[d][r] / lrow[r];
      Ob[((size_t)(b * kT) + qg) * kDim + h * kD + d * 16 + l16] = f2bf(val);
    }
  }
}

// ---------------------------------------------------------------------------
// Host-side launch
// ---------------------------------------------------------------------------
extern "C" void kernel_launch(void* const* d_in, const int* in_sizes, int n_in,
                              void* d_out, int out_size, void* d_ws, size_t ws_size,
                              hipStream_t stream) {
  const float* x  = (const float*)d_in[0];
  // d_in[1] = attn_mask (causal) — computed analytically in-kernel
  const float* Wq = (const float*)d_in[2];
  const float* bq = (const float*)d_in[3];
  const float* Wk = (const float*)d_in[4];
  const float* bk = (const float*)d_in[5];
  const float* Wv = (const float*)d_in[6];
  const float* bv = (const float*)d_in[7];
  const float* Wo = (const float*)d_in[8];
  const float* bo = (const float*)d_in[9];
  float* out = (float*)d_out;

  char* ws = (char*)d_ws;
  float*  q_f32 = (float*)ws;  ws += (size_t)kM * kDim * 4;   // 32 MB
  float*  k_f32 = (float*)ws;  ws += (size_t)kM * kD * 4;     //  2 MB
  float*  v_f32 = (float*)ws;  ws += (size_t)kM * kD * 4;     //  2 MB
  __bf16* qb    = (__bf16*)ws; ws += (size_t)kM * kDim * 2;   // 16 MB
  __bf16* kb    = (__bf16*)ws; ws += (size_t)kM * kD * 2;     //  1 MB
  __bf16* vb    = (__bf16*)ws; ws += (size_t)kM * kD * 2;     //  1 MB
  __bf16* ob    = (__bf16*)ws; ws += (size_t)kM * kDim * 2;   // 16 MB

  const dim3 blk(256);

  // Projections (x is f32; converted to bf16 during LDS staging)
  gemm_bias_kernel<false><<<dim3(kDim / 128, kM / 128), blk, 0, stream>>>(
      x, Wq, bq, q_f32, kM, kDim, kDim);
  gemm_bias_kernel<false><<<dim3(kD / 128, kM / 128), blk, 0, stream>>>(
      x, Wk, bk, k_f32, kM, kD, kDim);
  gemm_bias_kernel<false><<<dim3(kD / 128, kM / 128), blk, 0, stream>>>(
      x, Wv, bv, v_f32, kM, kD, kDim);

  // RoPE + bf16 conversion / relayout
  rope_q_kernel<<<(kB * kT * kH * 64) / 256, blk, 0, stream>>>(q_f32, qb);
  rope_k_kernel<<<(kB * kT * 64) / 256, blk, 0, stream>>>(k_f32, kb);
  cast_v_kernel<<<(kB * kT * kD) / 256, blk, 0, stream>>>(v_f32, vb);

  // Causal flash attention (MQA): grid = (q-tile, head, batch)
  flash_mqa_kernel<<<dim3(kT / 128, kH, kB), blk, 0, stream>>>(qb, kb, vb, ob);

  // Output projection
  gemm_bias_kernel<true><<<dim3(kDim / 128, kM / 128), blk, 0, stream>>>(
      ob, Wo, bo, out, kM, kDim, kDim);
}
